// simple_LSTM_78907139162239
// MI455X (gfx1250) — matrix operand order repaired and checked by reading, not compiled, not run on hardware
//
#include <hip/hip_runtime.h>

// ---------------------------------------------------------------------------
// Persistent batched-LSTM for MI455X (gfx1250, wave32, WMMA).
//  * 128 blocks x 416 threads (13 waves); each block owns 16 batch rows for
//    the whole T=512 recurrence (rows are independent -> no global sync).
//  * gates[16x416] = x[16x128]*W_ih^T + h[16x128]*W_hh^T via
//    v_wmma_f32_16x16x32_bf16; each wave owns 2 N-tiles (13*2*16 = 416).
//  * Weights converted once to bf16 WMMA B-fragments held in VGPRs.
//  * Activations stream through LDS (~49 KB); everything is L2-resident.
// ---------------------------------------------------------------------------

typedef __attribute__((ext_vector_type(16))) __bf16 v16bf;
typedef __attribute__((ext_vector_type(8)))  float  v8f;

#define B_TOT    2048
#define T_LEN    512
#define HID      100
#define DIN      100
#define G4       400          // 4*HID
#define NPAD     416          // 400 padded to 26 tiles of 16
#define KPAD     128          // 100 padded to 4 k-slices of 32
#define ROWS     16           // batch rows per block
#define NWAVES   13
#define NTHREADS (NWAVES * 32) // 416

union FragBF { v16bf bf; unsigned short us[16]; uint4 q[2]; };

__device__ __forceinline__ unsigned short f2bf(float f) {
  unsigned u = __float_as_uint(f);
  u += 0x7FFFu + ((u >> 16) & 1u);      // round to nearest even
  return (unsigned short)(u >> 16);
}

// A-matrix 16x32 bf16 fragment from LDS (row-major [16][KPAD]).
// ISA 7.12.2: lanes 0-15 : M=lane,    e0-7 -> K=+0..7,  e8-15 -> K=+16..23
//             lanes 16-31: M=lane-16, e0-7 -> K=+8..15, e8-15 -> K=+24..31
__device__ __forceinline__ v16bf load_a_frag(const unsigned short* base,
                                             int lane, int kt) {
  int m  = lane & 15;
  int kb = kt * 32 + ((lane < 16) ? 0 : 8);
  const unsigned short* p = base + m * KPAD + kb;
  FragBF a;
  a.q[0] = *(const uint4*)(p);        // 8 bf16, contiguous -> ds_load_b128
  a.q[1] = *(const uint4*)(p + 16);   // 8 bf16
  return a.bf;
}

__global__ __launch_bounds__(NTHREADS, 1)
void lstm_persistent(const int* __restrict__ x_in, const float* __restrict__ emb,
                     const float* __restrict__ W_ih, const float* __restrict__ W_hh,
                     const float* __restrict__ b_ih, const float* __restrict__ b_hh,
                     const float* __restrict__ W_dec, const float* __restrict__ b_dec,
                     const float* __restrict__ hx0,  const float* __restrict__ cx0,
                     float* __restrict__ out) {
  __shared__ __align__(16) unsigned short xA[ROWS][KPAD];   // x_t  (bf16)
  __shared__ __align__(16) unsigned short hA[ROWS][KPAD];   // h_t  (bf16)
  __shared__ __align__(16) float gates[ROWS][NPAD];
  __shared__ float cxs[ROWS][HID];
  __shared__ float hxs[ROWS][HID];
  __shared__ float biassum[NPAD];
  __shared__ int   toks[ROWS];

  const int tid  = threadIdx.x;
  const int lane = tid & 31;
  const int wid  = tid >> 5;          // wave id 0..12
  const int b0   = blockIdx.x * ROWS;

  // ---------------- one-time init ----------------
  for (int i = tid; i < NPAD; i += NTHREADS)
    biassum[i] = (i < G4) ? (b_ih[i] + b_hh[i]) : 0.f;
  for (int i = tid; i < ROWS * HID; i += NTHREADS) {
    int r = i / HID, k = i % HID;
    float h   = hx0[(b0 + r) * HID + k];
    hxs[r][k] = h;
    hA[r][k]  = f2bf(h);
    cxs[r][k] = cx0[(b0 + r) * HID + k];
  }
  for (int i = tid; i < ROWS * (KPAD - HID); i += NTHREADS) {
    int r = i / (KPAD - HID), k = HID + i % (KPAD - HID);
    hA[r][k] = 0; xA[r][k] = 0;       // K padding stays zero forever
  }

  // ------- weight B-fragments, register resident (bf16) -------
  // Wave `wid` owns n-tiles 2*wid and 2*wid+1.
  // B layout: N = lane&15; lanes 0-15 e0-7 -> K=+0..7, e8-15 -> K=+16..23; lanes 16-31 e0-7 -> K=+8..15, e8-15 -> K=+24..31.
  v16bf wihf[2][4], whhf[2][4];
  #pragma unroll
  for (int ntl = 0; ntl < 2; ++ntl) {
    int n = (2 * wid + ntl) * 16 + (lane & 15);
    #pragma unroll
    for (int kt = 0; kt < 4; ++kt) {
      FragBF fi, fh;
      #pragma unroll
      for (int e = 0; e < 16; ++e) {
        int k   = kt * 32 + ((lane < 16) ? 0 : 8) + ((e < 8) ? e : e + 8);
        bool ok = (n < G4) && (k < DIN);
        fi.us[e] = ok ? f2bf(W_ih[n * DIN + k]) : (unsigned short)0;
        fh.us[e] = ok ? f2bf(W_hh[n * HID + k]) : (unsigned short)0;
      }
      wihf[ntl][kt] = fi.bf;
      whhf[ntl][kt] = fh.bf;
    }
  }
  __syncthreads();

  // ---------------- recurrence over T ----------------
  for (int t = 0; t < T_LEN; ++t) {
    if (tid < ROWS) {
      toks[tid] = x_in[(b0 + tid) * T_LEN + t];
      if (t + 1 < T_LEN) {            // warm L2/WGP$ for next step's rows
        int tok1 = x_in[(b0 + tid) * T_LEN + t + 1];
        __builtin_prefetch(emb + (size_t)tok1 * DIN, 0, 3);
      }
    }
    __syncthreads();

    // embedding gather -> bf16 LDS tile
    for (int i = tid; i < ROWS * DIN; i += NTHREADS) {
      int r = i / DIN, k = i % DIN;
      xA[r][k] = f2bf(emb[(size_t)toks[r] * DIN + k]);
    }
    __syncthreads();

    // gates = xA @ W_ih^T + hA @ W_hh^T  (WMMA, f32 accum)
    #pragma unroll
    for (int ntl = 0; ntl < 2; ++ntl) {
      const int nt = 2 * wid + ntl;
      v8f acc = {0.f, 0.f, 0.f, 0.f, 0.f, 0.f, 0.f, 0.f};
      #pragma unroll
      for (int kt = 0; kt < 4; ++kt) {
        v16bf ax = load_a_frag(&xA[0][0], lane, kt);
        acc = __builtin_amdgcn_wmma_f32_16x16x32_bf16(
                  false, ax, false, wihf[ntl][kt], (short)0, acc, false, false);
        v16bf ah = load_a_frag(&hA[0][0], lane, kt);
        acc = __builtin_amdgcn_wmma_f32_16x16x32_bf16(
                  false, ah, false, whhf[ntl][kt], (short)0, acc, false, false);
      }
      // C layout: vgpr r -> M = r + (lane<16 ? 0 : 8), N = nt*16 + (lane&15)
      const int nn    = nt * 16 + (lane & 15);
      const int mbase = (lane < 16) ? 0 : 8;
      const float bs  = biassum[nn];
      #pragma unroll
      for (int r = 0; r < 8; ++r)
        gates[mbase + r][nn] = acc[r] + bs;
    }
    __syncthreads();

    // elementwise cell update (gate order i,f,g,o; extra tanh per reference)
    for (int i = tid; i < ROWS * HID; i += NTHREADS) {
      int r = i / HID, k = i % HID;
      float iv = gates[r][k];
      float fv = gates[r][HID + k];
      float gv = gates[r][2 * HID + k];
      float ov = gates[r][3 * HID + k];
      iv = 1.f / (1.f + __expf(-iv));
      fv = 1.f / (1.f + __expf(-fv));
      gv = tanhf(gv);
      ov = 1.f / (1.f + __expf(-ov));
      float c = fv * cxs[r][k] + iv * gv;
      float h = tanhf(ov * tanhf(c));
      cxs[r][k] = c;
      hxs[r][k] = h;
      hA[r][k]  = f2bf(h);
    }
    __syncthreads();
  }

  // ---------------- decode: out[b] = hx . W_dec + b_dec ----------------
  if (tid < ROWS) {
    float acc = b_dec[0];
    #pragma unroll 4
    for (int k = 0; k < HID; ++k) acc += hxs[tid][k] * W_dec[k];
    out[b0 + tid] = acc;
  }
}

extern "C" void kernel_launch(void* const* d_in, const int* in_sizes, int n_in,
                              void* d_out, int out_size, void* d_ws, size_t ws_size,
                              hipStream_t stream) {
  const int*   x_in  = (const int*)  d_in[0];
  const float* emb   = (const float*)d_in[1];
  const float* W_ih  = (const float*)d_in[2];
  const float* W_hh  = (const float*)d_in[3];
  const float* b_ih  = (const float*)d_in[4];
  const float* b_hh  = (const float*)d_in[5];
  const float* W_dec = (const float*)d_in[6];
  const float* b_dec = (const float*)d_in[7];
  const float* hx0   = (const float*)d_in[8];
  const float* cx0   = (const float*)d_in[9];
  float* out = (float*)d_out;

  dim3 grid(B_TOT / ROWS);   // 128 persistent blocks
  dim3 block(NTHREADS);      // 13 waves of 32
  hipLaunchKernelGGL(lstm_persistent, grid, block, 0, stream,
                     x_in, emb, W_ih, W_hh, b_ih, b_hh,
                     W_dec, b_dec, hx0, cx0, out);
}
